// BahdanauAttentionLSTM_70317204570134
// MI455X (gfx1250) — matrix operand order repaired and checked by reading, not compiled, not run on hardware
//
#include <hip/hip_runtime.h>
#include <hip/hip_bf16.h>

#define B_  32
#define S_  2048
#define H_  1024
#define SBLK 128

typedef __attribute__((ext_vector_type(16))) __bf16 v16bf;
typedef __attribute__((ext_vector_type(8)))  float  v8f;

union Frag {
    uint4 q[2];
    unsigned int u[8];
    v16bf v;
};

// pack two fp32 into one dword of two bf16
#if __has_builtin(__builtin_amdgcn_cvt_pk_bf16_f32)
__device__ __forceinline__ unsigned int pk_bf16(float lo, float hi) {
    typedef __attribute__((ext_vector_type(2))) __bf16 v2bf;
    union { v2bf v; unsigned int u; } cv;
    cv.v = __builtin_amdgcn_cvt_pk_bf16_f32(lo, hi);
    return cv.u;
}
#else
__device__ __forceinline__ unsigned int pk_bf16(float lo, float hi) {
    unsigned int a = __float_as_uint(lo);
    unsigned int b = __float_as_uint(hi);
    return (a >> 16) | (b & 0xFFFF0000u);   // truncate
}
#endif

#if __has_builtin(__builtin_amdgcn_tanhf)
#define FAST_TANH(x) __builtin_amdgcn_tanhf(x)
#else
#define FAST_TANH(x) tanhf(x)
#endif

// ---------------------------------------------------------------------------
// Kernel 1: t[b][o] = sum_h (q0+q1)[b][h] * Wa[o][h] + Wa_b[o] + Ua_b[o]
// ---------------------------------------------------------------------------
__global__ __launch_bounds__(256) void wq_kernel(
    const float* __restrict__ q0, const float* __restrict__ q1,
    const float* __restrict__ Wa, const float* __restrict__ Wa_b,
    const float* __restrict__ Ua_b, float* __restrict__ t_ws)
{
    int idx = blockIdx.x * 256 + threadIdx.x;      // 0 .. 32767
    int b = idx >> 10, o = idx & 1023;
    const float4* qa = (const float4*)(q0 + (size_t)b * H_);
    const float4* qb = (const float4*)(q1 + (size_t)b * H_);
    const float4* w  = (const float4*)(Wa + (size_t)o * H_);
    float acc = 0.f;
    #pragma unroll 4
    for (int h = 0; h < H_ / 4; ++h) {
        float4 x = qa[h], y = qb[h], z = w[h];
        acc += (x.x + y.x) * z.x + (x.y + y.y) * z.y +
               (x.z + y.z) * z.z + (x.w + y.w) * z.w;
    }
    t_ws[idx] = acc + Wa_b[o] + Ua_b[o];
}

// ---------------------------------------------------------------------------
// Kernel 2: scores[b][s] = sum_o Va[o]*tanh( (keys@Ua^T)[b][s][o] + t[b][o] )
// bf16 WMMA GEMM. keys tile (128x1024) staged once in LDS; Ua chunks staged
// through a double-buffered LDS stage (1 barrier / k-chunk, loads overlapped
// with WMMA). 8 waves x (16-row M-tile, 8 C accumulators = 128 cols/pass).
// ---------------------------------------------------------------------------
__global__ __launch_bounds__(256) void score_kernel(
    const float* __restrict__ keys, const float* __restrict__ Ua,
    const float* __restrict__ t_ws, const float* __restrict__ Va_w,
    const float* __restrict__ Va_b, float* __restrict__ scores)
{
    extern __shared__ char smem[];
    unsigned short* Alds = (unsigned short*)smem;            // 128 x 1032 halves (264192 B)
    unsigned short* Blds = (unsigned short*)(smem + 264192); // 2 x 128 x 40 halves (20480 B)
    float* Tlds = (float*)(smem + 284672);                   // 1024 f32
    float* Vlds = (float*)(smem + 288768);                   // 1024 f32

    const int tid  = threadIdx.x;
    const int wave = tid >> 5;
    const int lane = tid & 31;
    const int l16  = lane & 15;
    const int hk   = lane >> 4;
    const int b    = blockIdx.y;
    const int s0   = blockIdx.x * SBLK;

    // ---- stage keys tile (128 x 1024 fp32 -> bf16 in LDS, padded stride 1032)
    for (int it = tid; it < SBLK * 128; it += 256) {
        int row = it >> 7, seg = it & 127;
        const float4* src = (const float4*)(keys + ((size_t)b * S_ + s0 + row) * H_ + seg * 8);
        float4 x = src[0], y = src[1];
        uint4 o4;
        o4.x = pk_bf16(x.x, x.y); o4.y = pk_bf16(x.z, x.w);
        o4.z = pk_bf16(y.x, y.y); o4.w = pk_bf16(y.z, y.w);
        *(uint4*)(Alds + row * 1032 + seg * 8) = o4;
    }
    for (int o = tid; o < H_; o += 256) {
        Tlds[o] = t_ws[b * H_ + o];
        Vlds[o] = Va_w[o];
    }
    __syncthreads();

    float sp[8] = {0.f, 0.f, 0.f, 0.f, 0.f, 0.f, 0.f, 0.f};
    const unsigned short* Arow = Alds + (wave * 16 + l16) * 1032;

    const int r  = tid >> 1;     // Ua row (output column) this thread stages
    const int hf = tid & 1;      // which 16-element K half

    for (int op = 0; op < 8; ++op) {            // 8 passes of 128 output cols
        v8f c[8];
        #pragma unroll
        for (int j = 0; j < 8; ++j)
            c[j] = (v8f){0.f, 0.f, 0.f, 0.f, 0.f, 0.f, 0.f, 0.f};

        const float* Bbase = Ua + ((size_t)(op * 128 + r)) * H_ + hf * 16;

        // preload k-chunk 0 into registers
        float4 rg[4];
        {
            const float4* src = (const float4*)(Bbase);
            rg[0] = src[0]; rg[1] = src[1]; rg[2] = src[2]; rg[3] = src[3];
        }

        for (int kc = 0; kc < H_ / 32; ++kc) {
            const int k = kc * 32;
            unsigned short* Bb = Blds + (kc & 1) * 5120;

            // store current chunk (bf16) into this iteration's buffer
            {
                uint4 p0, p1;
                p0.x = pk_bf16(rg[0].x, rg[0].y); p0.y = pk_bf16(rg[0].z, rg[0].w);
                p0.z = pk_bf16(rg[1].x, rg[1].y); p0.w = pk_bf16(rg[1].z, rg[1].w);
                p1.x = pk_bf16(rg[2].x, rg[2].y); p1.y = pk_bf16(rg[2].z, rg[2].w);
                p1.z = pk_bf16(rg[3].x, rg[3].y); p1.w = pk_bf16(rg[3].z, rg[3].w);
                uint4* dst = (uint4*)(Bb + r * 40 + hf * 16);
                dst[0] = p0; dst[1] = p1;
            }
            // issue next chunk's global loads (consumed next iteration)
            if (kc + 1 < H_ / 32) {
                const float4* src = (const float4*)(Bbase + (kc + 1) * 32);
                rg[0] = src[0]; rg[1] = src[1]; rg[2] = src[2]; rg[3] = src[3];
            }
            __syncthreads();

            // A fragment: row M=l16, K contiguous {k+16*hk .. k+16*hk+15}
            Frag a;
            a.q[0] = *(const uint4*)(Arow + k + 16 * hk);
            a.q[1] = *(const uint4*)(Arow + k + 16 * hk + 8);

            // software-pipelined B fragments: col N=l16 of tile j
            const unsigned short* Bcol = Bb + l16 * 40 + 16 * hk;
            Frag bb;
            bb.q[0] = *(const uint4*)(Bcol);
            bb.q[1] = *(const uint4*)(Bcol + 8);
            #pragma unroll
            for (int j = 0; j < 8; ++j) {
                Frag bn = bb;
                if (j < 7) {
                    const unsigned short* Bp = Bcol + (j + 1) * 640; // 16 rows * 40
                    bn.q[0] = *(const uint4*)(Bp);
                    bn.q[1] = *(const uint4*)(Bp + 8);
                }
                c[j] = __builtin_amdgcn_wmma_f32_16x16x32_bf16(
                    false, a.v, false, bb.v, (short)0, c[j], false, false);
                bb = bn;
            }
        }

        // fuse tanh + Va dot into per-lane score partials
        #pragma unroll
        for (int j = 0; j < 8; ++j) {
            int o = op * 128 + j * 16 + l16;
            float tv = Tlds[o], vv = Vlds[o];
            #pragma unroll
            for (int i = 0; i < 8; ++i)
                sp[i] += vv * FAST_TANH(c[j][i] + tv);
        }
    }

    // reduce over N (16 lanes of each half; masks 1,2,4,8 stay in-half)
    #pragma unroll
    for (int i = 0; i < 8; ++i) {
        float v = sp[i];
        v += __shfl_xor(v, 1, 32);
        v += __shfl_xor(v, 2, 32);
        v += __shfl_xor(v, 4, 32);
        v += __shfl_xor(v, 8, 32);
        sp[i] = v;
    }
    if (l16 == 0) {
        float vb = Va_b[0];
        int srow = s0 + wave * 16 + hk * 8;   // rows i / i+8 per C layout
        #pragma unroll
        for (int i = 0; i < 8; ++i)
            scores[(size_t)b * S_ + srow + i] = sp[i] + vb;
    }
}

// ---------------------------------------------------------------------------
// Kernel 3: softmax over S per batch
// ---------------------------------------------------------------------------
__global__ __launch_bounds__(256) void softmax_kernel(
    const float* __restrict__ scores, float* __restrict__ wout)
{
    __shared__ float red[256];
    int b = blockIdx.x, tid = threadIdx.x;
    const float* sr = scores + (size_t)b * S_;
    float* wo = wout + (size_t)b * S_;

    float m = -3.4e38f;
    for (int s = tid; s < S_; s += 256) m = fmaxf(m, sr[s]);
    red[tid] = m; __syncthreads();
    for (int off = 128; off; off >>= 1) {
        if (tid < off) red[tid] = fmaxf(red[tid], red[tid + off]);
        __syncthreads();
    }
    m = red[0]; __syncthreads();

    float sum = 0.f;
    for (int s = tid; s < S_; s += 256) {
        float e = __expf(sr[s] - m);
        wo[s] = e;
        sum += e;
    }
    red[tid] = sum; __syncthreads();
    for (int off = 128; off; off >>= 1) {
        if (tid < off) red[tid] += red[tid + off];
        __syncthreads();
    }
    float inv = 1.f / red[0];
    for (int s = tid; s < S_; s += 256) wo[s] *= inv;
}

// ---------------------------------------------------------------------------
// Kernel 4: context[b][h] = sum_s weights[b][s] * keys[b][s][h]
// ---------------------------------------------------------------------------
__global__ __launch_bounds__(256) void context_kernel(
    const float* __restrict__ keys, const float* __restrict__ w,
    float* __restrict__ ctx)
{
    int b = blockIdx.x >> 2, qt = blockIdx.x & 3;
    int h = qt * 256 + threadIdx.x;
    const float* wr = w + (size_t)b * S_;
    const float* kp = keys + (size_t)b * S_ * H_ + h;
    float acc = 0.f;
    #pragma unroll 4
    for (int s = 0; s < S_; ++s)
        acc += wr[s] * kp[(size_t)s * H_];
    ctx[(size_t)b * H_ + h] = acc;
}

// ---------------------------------------------------------------------------
extern "C" void kernel_launch(void* const* d_in, const int* in_sizes, int n_in,
                              void* d_out, int out_size, void* d_ws, size_t ws_size,
                              hipStream_t stream)
{
    const float* q0   = (const float*)d_in[0];
    const float* q1   = (const float*)d_in[1];
    const float* keys = (const float*)d_in[2];
    const float* Wa_w = (const float*)d_in[3];
    const float* Wa_b = (const float*)d_in[4];
    const float* Ua_w = (const float*)d_in[5];
    const float* Ua_b = (const float*)d_in[6];
    const float* Va_w = (const float*)d_in[7];
    const float* Va_b = (const float*)d_in[8];

    float* out     = (float*)d_out;
    float* ctx     = out;                 // B*1*H = 32768 floats
    float* weights = out + B_ * H_;       // B*1*S = 65536 floats

    float* t_ws   = (float*)d_ws;         // B*H floats
    float* scores = t_ws + B_ * H_;       // B*S floats

    wq_kernel<<<(B_ * H_) / 256, 256, 0, stream>>>(q0, q1, Wa_w, Wa_b, Ua_b, t_ws);

    dim3 grid(S_ / SBLK, B_);             // 16 x 32 workgroups
    size_t smem = 264192 + 20480 + 4096 + 4096;   // 292864 B dynamic LDS
    score_kernel<<<grid, 256, smem, stream>>>(keys, Ua_w, t_ws, Va_w, Va_b, scores);

    softmax_kernel<<<B_, 256, 0, stream>>>(scores, weights);

    context_kernel<<<B_ * 4, 256, 0, stream>>>(keys, weights, ctx);
}
